// SigLIPBlock_14001593385467
// MI455X (gfx1250) — matrix operand order repaired and checked by reading, not compiled, not run on hardware
//
#include <hip/hip_runtime.h>
#include <hip/hip_bf16.h>
#include <math.h>

#define DEVFN __device__ __forceinline__

typedef __attribute__((ext_vector_type(16))) __bf16 v16bf;
typedef __attribute__((ext_vector_type(8)))  __bf16 v8bf;
typedef __attribute__((ext_vector_type(8)))  float  v8f;
typedef __attribute__((ext_vector_type(4)))  unsigned int v4u;
typedef __attribute__((ext_vector_type(8)))  int v8i;
typedef __attribute__((ext_vector_type(4)))  int v4i;

union BF16x16 { v16bf v; v8bf h[2]; __bf16 e[16]; };

constexpr int Bc = 8, Tc = 1024, Dc = 1152, NHc = 16, Fc = 4304, HDc = 72;
constexpr int NROW = Bc * Tc;

#ifndef HAVE_TDM
#if defined(__has_builtin)
#if __has_builtin(__builtin_amdgcn_tensor_load_to_lds)
#define HAVE_TDM 1
#endif
#endif
#endif
#ifndef HAVE_TDM
#define HAVE_TDM 0
#endif

DEVFN v8f wmma_bf16(v16bf a, v16bf b, v8f c) {
  // D = A(16x32 bf16) * B(32x16 bf16) + C(16x16 f32)
  return __builtin_amdgcn_wmma_f32_16x16x32_bf16(false, a, false, b, (short)0, c,
                                                 false, false);
}

#if HAVE_TDM
// Tensor Data Mover: async DMA of a (tile1 x tile0) bf16 tile from a row-major
// tensor (stride0 elems/row) into densely-packed LDS. OOB reads (beyond
// td0/td1, measured from gsrc) return zero -> free K-tail / head-dim padding.
// D# layout per CDNA5 ISA.
DEVFN void tdm_load_2d(unsigned lds_off, const void* gsrc,
                       unsigned td0, unsigned td1,
                       unsigned tile0, unsigned tile1,
                       unsigned long stride0) {
  const unsigned long ga = (unsigned long)gsrc;
  v4u g0;
  g0[0] = 1u;                                        // count=1 (valid descr)
  g0[1] = lds_off;                                   // LDS byte address
  g0[2] = (unsigned)(ga & 0xffffffffu);              // global_addr[31:0]
  g0[3] = (unsigned)((ga >> 32) & 0x01ffffffu) | 0x80000000u; // [56:32]|type=2
  v8i g1;
  g1[0] = (int)(1u << 16);                           // wg_mask=0, data_size=1 (2B)
  g1[1] = (int)((td0 & 0xffffu) << 16);              // tensor_dim0[15:0]
  g1[2] = (int)(((td0 >> 16) & 0xffffu) | ((td1 & 0xffffu) << 16));
  g1[3] = (int)(((td1 >> 16) & 0xffffu) | ((tile0 & 0xffffu) << 16));
  g1[4] = (int)(tile1 & 0xffffu);                    // tile_dim1, tile_dim2=0
  g1[5] = (int)(unsigned)(stride0 & 0xffffffffu);    // dim0_stride[31:0]
  g1[6] = (int)(unsigned)((stride0 >> 32) & 0xffffu);// dim0_stride[47:32]
  g1[7] = 0;                                         // dim1_stride=0 (2D)
  const v4i z = {0, 0, 0, 0};
#if __clang_major__ >= 23
  const v8i z8 = {0, 0, 0, 0, 0, 0, 0, 0};
  __builtin_amdgcn_tensor_load_to_lds(g0, g1, z, z, z8, 0);
#else
  __builtin_amdgcn_tensor_load_to_lds(g0, g1, z, z, 0);
#endif
}
#else
// Fallback: cooperative zero-padded tile load by all 256 threads.
DEVFN void coop_load_tile(__bf16* dst, const __bf16* src,
                          unsigned td0, unsigned td1,
                          unsigned tile0, unsigned tile1, unsigned stride0) {
  const unsigned total = tile0 * tile1;
  for (unsigned idx = threadIdx.x; idx < total; idx += 256) {
    const unsigned r = idx / tile0, c = idx % tile0;
    dst[idx] = (r < td1 && c < td0) ? src[(size_t)r * stride0 + c] : (__bf16)0.0f;
  }
}
#endif

DEVFN void tile_fence(int wave) {
#if HAVE_TDM
  if (wave == 0) __builtin_amdgcn_s_wait_tensorcnt(0);
#endif
  __syncthreads();
  (void)wave;
}

// A-fragment (16x32 bf16) from global: lane<16 row m=lane, K {k0..+7, k0+16..+23};
// lane>=16 same row set, K shifted +8 (ISA 7.12.2).
DEVFN v16bf frag_a_g(const __bf16* rowptr, int k0, int Klim) {
  BF16x16 f;
  const int lane = threadIdx.x & 31;
  const int kb = k0 + ((lane & 16) ? 8 : 0);
  if (rowptr && (kb + 23) < Klim) {
    f.h[0] = *(const v8bf*)(rowptr + kb);
    f.h[1] = *(const v8bf*)(rowptr + kb + 16);
  } else {
#pragma unroll
    for (int j = 0; j < 8; ++j) {
      const int k1 = kb + j, k2 = kb + 16 + j;
      f.e[j]     = (rowptr && k1 < Klim) ? rowptr[k1] : (__bf16)0.0f;
      f.e[8 + j] = (rowptr && k2 < Klim) ? rowptr[k2] : (__bf16)0.0f;
    }
  }
  return f.v;
}

DEVFN float gelu_tanh(float x) {
  const float x3 = x * x * x;
  return 0.5f * x * (1.0f + tanhf(0.7978845608028654f * (x + 0.044715f * x3)));
}

enum { EPI_BF16 = 0, EPI_RES_F32 = 1, EPI_GELU_BF16 = 2 };

// C[M,N] = A[M,K](bf16) * W[N,K](bf16)^T + bias, fp32 accum.
// Block = 8 waves; block tile 64x64; wave (wm,wn) owns 16x32.
// TDM double-buffers 64x32 A/B tiles through LDS, overlapped with WMMA.
template <int EPI>
__global__ __launch_bounds__(256) void gemm_wmma_kernel(
    const __bf16* __restrict__ A, const __bf16* __restrict__ W,
    const float* __restrict__ bias, const float* __restrict__ res,
    float* __restrict__ outF, __bf16* __restrict__ outB,
    int M, int N, int K)
{
  __shared__ __bf16 sA[2][64 * 32];
  __shared__ __bf16 sB[2][64 * 32];

  const int wave = threadIdx.x >> 5, lane = threadIdx.x & 31;
  const int wm = wave & 3, wn = wave >> 2;
  const int m0 = blockIdx.y * 64;
  const int n0 = blockIdx.x * 64;

  auto load_tiles = [&](int buf, int k0) {
#if HAVE_TDM
    if (wave == 0) {
      tdm_load_2d((unsigned)(size_t)&sA[buf][0], A + (size_t)m0 * K + k0,
                  (unsigned)(K - k0), (unsigned)(M - m0), 32u, 64u,
                  (unsigned long)K);
      tdm_load_2d((unsigned)(size_t)&sB[buf][0], W + (size_t)n0 * K + k0,
                  (unsigned)(K - k0), (unsigned)(N - n0), 32u, 64u,
                  (unsigned long)K);
    }
#else
    coop_load_tile(&sA[buf][0], A + (size_t)m0 * K + k0,
                   (unsigned)(K - k0), (unsigned)(M - m0), 32u, 64u, (unsigned)K);
    coop_load_tile(&sB[buf][0], W + (size_t)n0 * K + k0,
                   (unsigned)(K - k0), (unsigned)(N - n0), 32u, 64u, (unsigned)K);
#endif
  };

  v8f acc0 = {}, acc1 = {};
  load_tiles(0, 0);
  tile_fence(wave);

  int buf = 0;
  for (int k0 = 0; k0 < K; k0 += 32) {
    if (k0 + 32 < K) load_tiles(buf ^ 1, k0 + 32);  // async next tile

    BF16x16 af, b0f, b1f;
    const int ar = (wm * 16 + (lane & 15)) * 32 + ((lane & 16) ? 8 : 0);
    af.h[0] = *(const v8bf*)&sA[buf][ar];
    af.h[1] = *(const v8bf*)&sA[buf][ar + 16];
    const int nl0 = (wn * 32 + (lane & 15)) * 32 + ((lane & 16) ? 8 : 0);
    b0f.h[0] = *(const v8bf*)&sB[buf][nl0];
    b0f.h[1] = *(const v8bf*)&sB[buf][nl0 + 16];
    const int nl1 = nl0 + 16 * 32;
    b1f.h[0] = *(const v8bf*)&sB[buf][nl1];
    b1f.h[1] = *(const v8bf*)&sB[buf][nl1 + 16];

    acc0 = wmma_bf16(af.v, b0f.v, acc0);
    acc1 = wmma_bf16(af.v, b1f.v, acc1);

    tile_fence(wave);   // next tile landed; everyone done with current
    buf ^= 1;
  }

  // C/D layout: VGPR r -> row (+r, +8 for lanes>=16); col lane%16 (+16*t).
  const int mb = m0 + wm * 16 + ((lane & 16) ? 8 : 0);
#pragma unroll
  for (int t = 0; t < 2; ++t) {
    const int n = n0 + wn * 32 + (lane & 15) + 16 * t;
    if (n >= N) continue;
    const float bval = bias ? bias[n] : 0.0f;
#pragma unroll
    for (int r = 0; r < 8; ++r) {
      const int m = mb + r;
      if (m >= M) continue;
      const float v = (t == 0 ? acc0[r] : acc1[r]) + bval;
      const size_t idx = (size_t)m * N + n;
      if (EPI == EPI_RES_F32)        outF[idx] = v + res[idx];
      else if (EPI == EPI_GELU_BF16) outB[idx] = (__bf16)gelu_tanh(v);
      else                           outB[idx] = (__bf16)v;
    }
  }
}

__global__ __launch_bounds__(256) void layernorm_bf16_kernel(
    const float* __restrict__ x, const float* __restrict__ w,
    const float* __restrict__ b, __bf16* __restrict__ out, int cols)
{
  const int row = blockIdx.x;
  const float* xr = x + (size_t)row * cols;
  float s = 0.f, s2 = 0.f;
  for (int c = threadIdx.x; c < cols; c += 256) {
    const float v = xr[c]; s += v; s2 += v * v;
  }
#pragma unroll
  for (int off = 16; off > 0; off >>= 1) {
    s  += __shfl_down(s,  off, 32);
    s2 += __shfl_down(s2, off, 32);
  }
  __shared__ float ps[8], ps2[8];
  __shared__ float mmean, mrstd;
  if ((threadIdx.x & 31) == 0) { ps[threadIdx.x >> 5] = s; ps2[threadIdx.x >> 5] = s2; }
  __syncthreads();
  if (threadIdx.x == 0) {
    float a = 0.f, a2 = 0.f;
    for (int i = 0; i < 8; ++i) { a += ps[i]; a2 += ps2[i]; }
    const float mean = a / cols;
    const float var  = a2 / cols - mean * mean;
    mmean = mean; mrstd = rsqrtf(var + 1e-5f);
  }
  __syncthreads();
  const float mean = mmean, rstd = mrstd;
  for (int c = threadIdx.x; c < cols; c += 256)
    out[(size_t)row * cols + c] = (__bf16)(((xr[c] - mean) * rstd) * w[c] + b[c]);
}

__global__ void cvt_f32_bf16_kernel(const float* __restrict__ in,
                                    __bf16* __restrict__ out, size_t n) {
  const size_t i = (size_t)blockIdx.x * blockDim.x + threadIdx.x;
  if (i < n) out[i] = (__bf16)in[i];
}

// Flash attention. Heads are contiguous [T,72] slabs (reference reshapes
// [B,T,D]->[B,NH,T,HD] with no transpose). Grid: (B*NH, T/128); 8 waves x 16
// query rows. K/V tiles (32 keys x 96 cols, zero-padded past 72 by TDM OOB)
// double-buffered through LDS.
__global__ __launch_bounds__(256) void attn_flash_kernel(
    const __bf16* __restrict__ Q, const __bf16* __restrict__ Kmat,
    const __bf16* __restrict__ V, __bf16* __restrict__ O)
{
  __shared__ __bf16 sK[2][32 * 96];
  __shared__ __bf16 sV[2][32 * 96];
  __shared__ __bf16 sP[8][16][32];

  const int wave = threadIdx.x >> 5, lane = threadIdx.x & 31;
  const int bh = blockIdx.x;
  const int b = bh / NHc, h = bh % NHc;
  const size_t base = (size_t)b * Tc * Dc + (size_t)h * Tc * HDc;
  const __bf16* Qh = Q + base;
  const __bf16* Kh = Kmat + base;
  const __bf16* Vh = V + base;

  const int q0 = blockIdx.y * 128 + wave * 16;
  const int qrow = q0 + (lane & 15);

  v16bf qf[3];
#pragma unroll
  for (int c = 0; c < 3; ++c)
    qf[c] = frag_a_g(Qh + (size_t)qrow * HDc, c * 32, HDc);  // K=72 zero-padded

  float mrun[8], lrun[8];
  v8f oacc[5];
  const v8f vzero = {};
#pragma unroll
  for (int r = 0; r < 8; ++r) { mrun[r] = -3.0e38f; lrun[r] = 0.f; }
#pragma unroll
  for (int t = 0; t < 5; ++t) oacc[t] = vzero;

  const float scale = 0.11785113019775793f;  // 72^-0.5

  auto load_kv = [&](int buf, int s0) {
#if HAVE_TDM
    if (wave == 0) {
      tdm_load_2d((unsigned)(size_t)&sK[buf][0], Kh + (size_t)s0 * HDc,
                  (unsigned)HDc, (unsigned)(Tc - s0), 96u, 32u,
                  (unsigned long)HDc);
      tdm_load_2d((unsigned)(size_t)&sV[buf][0], Vh + (size_t)s0 * HDc,
                  (unsigned)HDc, (unsigned)(Tc - s0), 96u, 32u,
                  (unsigned long)HDc);
    }
#else
    coop_load_tile(&sK[buf][0], Kh + (size_t)s0 * HDc,
                   (unsigned)HDc, (unsigned)(Tc - s0), 96u, 32u, (unsigned)HDc);
    coop_load_tile(&sV[buf][0], Vh + (size_t)s0 * HDc,
                   (unsigned)HDc, (unsigned)(Tc - s0), 96u, 32u, (unsigned)HDc);
#endif
  };

  load_kv(0, 0);
  tile_fence(wave);

  int buf = 0;
  for (int s0 = 0; s0 < Tc; s0 += 32) {
    if (s0 + 32 < Tc) load_kv(buf ^ 1, s0 + 32);  // async next K/V tile

    // S(16x32) = Q(16x72) * K^T : two 16x16 WMMA tiles over 3 K-chunks.
    v8f sa0 = {}, sa1 = {};
#pragma unroll
    for (int c = 0; c < 3; ++c) {
      const int kb = c * 32 + ((lane & 16) ? 8 : 0);   // 0..72, pad cols = 0
      const int key0 = lane & 15;
      BF16x16 b0, b1;
      b0.h[0] = *(const v8bf*)&sK[buf][key0 * 96 + kb];
      b0.h[1] = *(const v8bf*)&sK[buf][key0 * 96 + kb + 16];
      b1.h[0] = *(const v8bf*)&sK[buf][(key0 + 16) * 96 + kb];
      b1.h[1] = *(const v8bf*)&sK[buf][(key0 + 16) * 96 + kb + 16];
      sa0 = wmma_bf16(qf[c], b0.v, sa0);
      sa1 = wmma_bf16(qf[c], b1.v, sa1);
    }

    // Online softmax across this tile's 32 keys (row groups of 16 lanes).
    float corr[8];
#pragma unroll
    for (int r = 0; r < 8; ++r) {
      const float v0 = sa0[r] * scale;
      const float v1 = sa1[r] * scale;
      float mx = fmaxf(v0, v1);
#pragma unroll
      for (int off = 1; off < 16; off <<= 1) mx = fmaxf(mx, __shfl_xor(mx, off, 32));
      const float mnew = fmaxf(mrun[r], mx);
      const float p0 = __expf(v0 - mnew);
      const float p1 = __expf(v1 - mnew);
      const float cr = __expf(mrun[r] - mnew);
      float psum = p0 + p1;
#pragma unroll
      for (int off = 1; off < 16; off <<= 1) psum += __shfl_xor(psum, off, 32);
      lrun[r] = lrun[r] * cr + psum;
      mrun[r] = mnew;
      corr[r] = cr;
      const int m = r + ((lane & 16) ? 8 : 0);
      sP[wave][m][lane & 15]        = (__bf16)p0;
      sP[wave][m][(lane & 15) + 16] = (__bf16)p1;
    }
#pragma unroll
    for (int t = 0; t < 5; ++t)
#pragma unroll
      for (int r = 0; r < 8; ++r) oacc[t][r] *= corr[r];

    // Re-read P in A-fragment layout (own wave's region; DS ops in-order).
    BF16x16 pf;
    {
      const int m = lane & 15;
      const int kofs = (lane & 16) ? 8 : 0;
      pf.h[0] = *(const v8bf*)&sP[wave][m][kofs];
      pf.h[1] = *(const v8bf*)&sP[wave][m][kofs + 16];
    }
    // O(16x72) += P(16x32) * V(32x72): 5 N-tiles (last half-padded).
#pragma unroll
    for (int t = 0; t < 5; ++t) {
      BF16x16 vf;
      const int hd = t * 16 + (lane & 15);   // < 96; pad cols hold zeros
      const int kb = (lane & 16) ? 8 : 0;
#pragma unroll
      for (int j = 0; j < 16; ++j) vf.e[j] = sV[buf][(kb + j + (j >= 8 ? 8 : 0)) * 96 + hd];
      oacc[t] = wmma_bf16(pf.v, vf.v, oacc[t]);
    }

    tile_fence(wave);
    buf ^= 1;
  }

  // Normalize, store with the transpose(1,2)+merge layout: [B,T,NH*HD].
#pragma unroll
  for (int t = 0; t < 5; ++t) {
    const int hd = t * 16 + (lane & 15);
    if (hd >= HDc) continue;
#pragma unroll
    for (int r = 0; r < 8; ++r) {
      const int row = q0 + r + ((lane & 16) ? 8 : 0);
      O[(size_t)b * Tc * Dc + (size_t)row * Dc + h * HDc + hd] =
          (__bf16)(oacc[t][r] / lrun[r]);
    }
  }
}

extern "C" void kernel_launch(void* const* d_in, const int* in_sizes, int n_in,
                              void* d_out, int out_size, void* d_ws, size_t ws_size,
                              hipStream_t stream) {
  (void)in_sizes; (void)n_in; (void)out_size; (void)ws_size;
  const float* x    = (const float*)d_in[0];
  const float* ln1w = (const float*)d_in[1];
  const float* ln1b = (const float*)d_in[2];
  const float* wq   = (const float*)d_in[3];
  const float* bq   = (const float*)d_in[4];
  const float* wk   = (const float*)d_in[5];
  const float* bk   = (const float*)d_in[6];
  const float* wv   = (const float*)d_in[7];
  const float* bv   = (const float*)d_in[8];
  const float* wo   = (const float*)d_in[9];
  const float* bo   = (const float*)d_in[10];
  const float* ln2w = (const float*)d_in[11];
  const float* ln2b = (const float*)d_in[12];
  const float* w1   = (const float*)d_in[13];
  const float* b1   = (const float*)d_in[14];
  const float* w2   = (const float*)d_in[15];
  const float* b2   = (const float*)d_in[16];
  float* out = (float*)d_out;

  char* p = (char*)d_ws;
  auto alloc = [&](size_t bytes) -> char* {
    char* r = p; p += (bytes + 255) & ~(size_t)255; return r;
  };
  __bf16* h1  = (__bf16*)alloc((size_t)NROW * Dc * 2);
  __bf16* qb  = (__bf16*)alloc((size_t)NROW * Dc * 2);
  __bf16* kb_ = (__bf16*)alloc((size_t)NROW * Dc * 2);
  __bf16* vb  = (__bf16*)alloc((size_t)NROW * Dc * 2);
  __bf16* ab  = (__bf16*)alloc((size_t)NROW * Dc * 2);
  float*  x1  = (float*) alloc((size_t)NROW * Dc * 4);
  __bf16* h2  = (__bf16*)alloc((size_t)NROW * Dc * 2);
  __bf16* fb  = (__bf16*)alloc((size_t)NROW * Fc * 2);
  __bf16* wqb = (__bf16*)alloc((size_t)Dc * Dc * 2);
  __bf16* wkb = (__bf16*)alloc((size_t)Dc * Dc * 2);
  __bf16* wvb = (__bf16*)alloc((size_t)Dc * Dc * 2);
  __bf16* wob = (__bf16*)alloc((size_t)Dc * Dc * 2);
  __bf16* w1b = (__bf16*)alloc((size_t)Fc * Dc * 2);
  __bf16* w2b = (__bf16*)alloc((size_t)Dc * Fc * 2);

  auto cvt = [&](const float* src, __bf16* dst, size_t n) {
    cvt_f32_bf16_kernel<<<dim3((unsigned)((n + 255) / 256)), dim3(256), 0, stream>>>(src, dst, n);
  };
  cvt(wq, wqb, (size_t)Dc * Dc);
  cvt(wk, wkb, (size_t)Dc * Dc);
  cvt(wv, wvb, (size_t)Dc * Dc);
  cvt(wo, wob, (size_t)Dc * Dc);
  cvt(w1, w1b, (size_t)Fc * Dc);
  cvt(w2, w2b, (size_t)Dc * Fc);

  layernorm_bf16_kernel<<<dim3(NROW), dim3(256), 0, stream>>>(x, ln1w, ln1b, h1, Dc);

  const dim3 blk(256);
  const dim3 gD((Dc + 63) / 64, (NROW + 63) / 64);
  gemm_wmma_kernel<EPI_BF16><<<gD, blk, 0, stream>>>(h1, wqb, bq, nullptr, nullptr, qb,  NROW, Dc, Dc);
  gemm_wmma_kernel<EPI_BF16><<<gD, blk, 0, stream>>>(h1, wkb, bk, nullptr, nullptr, kb_, NROW, Dc, Dc);
  gemm_wmma_kernel<EPI_BF16><<<gD, blk, 0, stream>>>(h1, wvb, bv, nullptr, nullptr, vb,  NROW, Dc, Dc);

  attn_flash_kernel<<<dim3(Bc * NHc, Tc / 128), blk, 0, stream>>>(qb, kb_, vb, ab);

  gemm_wmma_kernel<EPI_RES_F32><<<gD, blk, 0, stream>>>(ab, wob, bo, x, x1, nullptr, NROW, Dc, Dc);

  layernorm_bf16_kernel<<<dim3(NROW), dim3(256), 0, stream>>>(x1, ln2w, ln2b, h2, Dc);

  const dim3 gF((Fc + 63) / 64, (NROW + 63) / 64);
  gemm_wmma_kernel<EPI_GELU_BF16><<<gF, blk, 0, stream>>>(h2, w1b, b1, nullptr, nullptr, fb, NROW, Fc, Dc);

  gemm_wmma_kernel<EPI_RES_F32><<<gD, blk, 0, stream>>>(fb, w2b, b2, x1, out, nullptr, NROW, Dc, Fc);
}
